// TransformerLayer_2293512536360
// MI455X (gfx1250) — compile-verified, matrix operand order checked by reading, not run on hardware
//
#include <hip/hip_runtime.h>
#include <hip/hip_fp16.h>

typedef __attribute__((ext_vector_type(16))) _Float16 v16h;
typedef __attribute__((ext_vector_type(8)))  float    v8f;

__device__ __forceinline__ v8f vzero8() {
    v8f z = {0.f, 0.f, 0.f, 0.f, 0.f, 0.f, 0.f, 0.f};
    return z;
}

// 16-bit A/B fragment K-offset swizzle (CDNA5 ISA 7.12.2, 16-bit A 16x32):
// VGPR pair p (0..7) -> K base offset; hi = lane>>4 selects the 8-wide K group.
__device__ __forceinline__ int frag_k(int p, int hi) {
    return (p < 4) ? (2 * p + 8 * hi) : (16 + 2 * (p - 4) + 8 * hi);
}

__device__ __forceinline__ v8f wmma16(v16h a, v16h b, v8f c) {
    return __builtin_amdgcn_wmma_f32_16x16x32_f16(
        /*neg_a=*/false, a, /*neg_b=*/false, b,
        /*c_mod=*/(short)0, c, /*reuse_a=*/false, /*reuse_b=*/false);
}

// ---- gfx1250 async global->LDS copies (ASYNCcnt-tracked, no VGPR round trip)
// Low 32 bits of a flat shared-aperture address are the LDS byte offset.
__device__ __forceinline__ unsigned lds_addr32(const void* p) {
    return (unsigned)(unsigned long long)p;
}
__device__ __forceinline__ void async_copy_b128(void* lds_dst, const void* gsrc) {
    asm volatile("global_load_async_to_lds_b128 %0, %1, off"
                 :: "v"(lds_addr32(lds_dst)), "v"(gsrc) : "memory");
}
__device__ __forceinline__ void async_copy_b64(void* lds_dst, const void* gsrc) {
    asm volatile("global_load_async_to_lds_b64 %0, %1, off"
                 :: "v"(lds_addr32(lds_dst)), "v"(gsrc) : "memory");
}
__device__ __forceinline__ void wait_async0() {
    asm volatile("s_wait_asynccnt 0x0" ::: "memory");
}

// ---------------------------------------------------------------- converts
__global__ void f32_to_f16_kernel(const float* __restrict__ src,
                                  _Float16* __restrict__ dst, int n) {
    int i = blockIdx.x * blockDim.x + threadIdx.x;
    if (i < n) dst[i] = (_Float16)src[i];
}

// ---------------------------------------------------------------- inorm
// InstanceNorm over D=512 per token (biased variance, eps=1e-5), f32 -> f16
__global__ void __launch_bounds__(256) inorm512_kernel(const float* __restrict__ x,
                                                       _Float16* __restrict__ y) {
    const int D = 512;
    int t = blockIdx.x;
    int tid = threadIdx.x;
    const float* row = x + (size_t)t * D;
    float a = row[tid], b = row[tid + 256];
    float s = a + b, ss = a * a + b * b;
#pragma unroll
    for (int o = 16; o > 0; o >>= 1) {
        s  += __shfl_xor(s, o, 32);
        ss += __shfl_xor(ss, o, 32);
    }
    __shared__ float rs[8], rss[8];
    __shared__ float meanv, rstdv;
    int wave = tid >> 5, lane = tid & 31;
    if (lane == 0) { rs[wave] = s; rss[wave] = ss; }
    __syncthreads();
    if (tid == 0) {
        float S1 = 0.f, S2 = 0.f;
#pragma unroll
        for (int i = 0; i < 8; ++i) { S1 += rs[i]; S2 += rss[i]; }
        float m = S1 / (float)D;
        float v = S2 / (float)D - m * m;
        meanv = m;
        rstdv = rsqrtf(v + 1e-5f);
    }
    __syncthreads();
    float m = meanv, r = rstdv;
    _Float16* yr = y + (size_t)t * D;
    yr[tid]       = (_Float16)((a - m) * r);
    yr[tid + 256] = (_Float16)((b - m) * r);
}

// ---------------------------------------------------------------- GEMM
// Y[M,N] = act((X[M,K] @ W[N,K]^T + bias[N]) * alpha)  (+ residual -> f32 out)
// Block = 128 threads (4 waves). Wave w owns 16x16 tile at (m0 + 16w, n0).
// A/B tiles staged with global_load_async_to_lds (ASYNCcnt path).
template <int RELU, int HAS_RES>
__global__ void __launch_bounds__(128)
gemm_xwt_kernel(const _Float16* __restrict__ X, const _Float16* __restrict__ W,
                const float* __restrict__ bias, const float* __restrict__ residual,
                float* __restrict__ Yf, _Float16* __restrict__ Yh,
                int M, int N, int K, float alpha) {
    __shared__ _Float16 Bs[16][40];       // W tile, [n][k]; 80B row stride (16B aligned)
    __shared__ _Float16 As[4][16][40];    // per-wave X tile, [m][k]
    int tid = threadIdx.x;
    int w = tid >> 5, lane = tid & 31;
    int hi = lane >> 4, lm = lane & 15;
    int n0 = blockIdx.x * 16;
    int m0 = blockIdx.y * 64 + w * 16;

    int brow = (tid * 4) >> 5, bcol = (tid * 4) & 31;   // B stage: 8B per thread
    int arow = lane >> 1, acol = (lane & 1) * 16;       // A stage: 32B per lane

    v8f acc = vzero8();
    for (int k0 = 0; k0 < K; k0 += 32) {
        // async-stage B: Bs[n][k] = W[n0+n][k0+k]
        async_copy_b64(&Bs[brow][bcol], W + (size_t)(n0 + brow) * K + k0 + bcol);
        // async-stage A: As[w][m][k] = X[m0+m][k0+k]
        const _Float16* xp = X + (size_t)(m0 + arow) * K + k0 + acol;
        async_copy_b128(&As[w][arow][acol], xp);
        async_copy_b128(&As[w][arow][acol + 8], xp + 8);
        if (k0 + 32 < K) __builtin_prefetch(xp + 32, 0, 1);
        wait_async0();
        __syncthreads();

        v16h af, bf;
#pragma unroll
        for (int p = 0; p < 8; ++p) {
            int kk = frag_k(p, hi);
            af[2 * p]     = As[w][lm][kk];
            af[2 * p + 1] = As[w][lm][kk + 1];
            bf[2 * p]     = Bs[lm][kk];
            bf[2 * p + 1] = Bs[lm][kk + 1];
        }
        acc = wmma16(af, bf, acc);
        __syncthreads();
    }
    int n = n0 + lm;
    float bv = bias[n];
#pragma unroll
    for (int r = 0; r < 8; ++r) {
        int m = m0 + r + 8 * hi;
        float v = (acc[r] + bv) * alpha;
        if (RELU) v = v > 0.f ? v : 0.f;
        size_t off = (size_t)m * N + n;
        if (HAS_RES) Yf[off] = residual[off] + v;
        else         Yh[off] = (_Float16)v;
    }
}

// ---------------------------------------------------------------- flash attn
// One wave per (16-query tile, head, batch). Head dim = 128, D = 512.
// Q pre-scaled by 1/sqrt(128). SELF: mask = g[b][q] & g[b][k] (g is (B,Sq) bytes).
// !SELF: mask is dense (B, Sq, Skv) bytes.
// K/V blocks staged with global_load_async_to_lds_b128; next block prefetched.
template <int SELF>
__global__ void __launch_bounds__(32)
flash_attn_kernel(const _Float16* __restrict__ Q, const _Float16* __restrict__ Kb,
                  const _Float16* __restrict__ Vb, const unsigned char* __restrict__ mask,
                  _Float16* __restrict__ O, int Sq, int Skv) {
    const int Dm = 512, DSd = 128;
    int b = blockIdx.z, h = blockIdx.y, q0 = blockIdx.x * 16;
    int lane = threadIdx.x, hi = lane >> 4, lm = lane & 15;

    __shared__ _Float16 Ks[32][136];   // [key][dim]; 272B row stride (16B aligned)
    __shared__ _Float16 Vs[32][136];
    __shared__ _Float16 Ps[16][36];    // P redistribution tile
    __shared__ unsigned char mq[16];
    __shared__ unsigned char mk[32];

    const _Float16* Qb  = Q  + ((size_t)b * Sq) * Dm + (size_t)h * DSd;
    const _Float16* Kbb = Kb + ((size_t)b * Skv) * Dm + (size_t)h * DSd;
    const _Float16* Vbb = Vb + ((size_t)b * Skv) * Dm + (size_t)h * DSd;

    // Q fragments in registers: 4 chunks of K=32 covering head dim 128
    v16h qf[4];
#pragma unroll
    for (int c = 0; c < 4; ++c) {
        const _Float16* qr = Qb + (size_t)lm * Dm + c * 32;
#pragma unroll
        for (int p = 0; p < 8; ++p) {
            int kk = frag_k(p, hi);
            qf[c][2 * p]     = qr[kk];
            qf[c][2 * p + 1] = qr[kk + 1];
        }
    }
    if (SELF) {
        if (lane < 16) mq[lane] = mask[(size_t)b * Sq + q0 + lane];
    }

    float mrow[8], lrow[8];
    v8f o[8];
#pragma unroll
    for (int r = 0; r < 8; ++r) { mrow[r] = -3.0e38f; lrow[r] = 0.f; o[r] = vzero8(); }

    for (int j0 = 0; j0 < Skv; j0 += 32) {
        {   // async-stage 32 K rows + 32 V rows (one row per lane, 8 x 16B at 32B stride)
            const _Float16* kr = Kbb + (size_t)(j0 + lane) * Dm;
            const _Float16* vr = Vbb + (size_t)(j0 + lane) * Dm;
#pragma unroll
            for (int c = 0; c < 8; ++c) {
                async_copy_b128(&Ks[lane][c * 16], kr + c * 16);
                async_copy_b128(&Vs[lane][c * 16], vr + c * 16);
            }
            if (j0 + 32 < Skv) {   // prefetch next block into cache hierarchy
                __builtin_prefetch(kr + 32 * Dm, 0, 1);
                __builtin_prefetch(vr + 32 * Dm, 0, 1);
            }
            if (SELF) mk[lane] = mask[(size_t)b * Skv + j0 + lane];
            wait_async0();
        }
        __syncthreads();

        // scores: two 16x16 tiles (keys j0..j0+15, j0+16..j0+31), K-dim 128
        v8f sc[2];
#pragma unroll
        for (int nt = 0; nt < 2; ++nt) {
            v8f a = vzero8();
#pragma unroll
            for (int c = 0; c < 4; ++c) {
                v16h bf;
                const _Float16* krow = &Ks[nt * 16 + lm][c * 32];
#pragma unroll
                for (int p = 0; p < 8; ++p) {
                    int kk = frag_k(p, hi);
                    bf[2 * p]     = krow[kk];
                    bf[2 * p + 1] = krow[kk + 1];
                }
                a = wmma16(qf[c], bf, a);
            }
            sc[nt] = a;
        }

        // mask (reference semantics: masked scores := -1e9)
#pragma unroll
        for (int nt = 0; nt < 2; ++nt) {
#pragma unroll
            for (int r = 0; r < 8; ++r) {
                bool ok;
                if (SELF) {
                    ok = (mq[r + 8 * hi] != 0) && (mk[nt * 16 + lm] != 0);
                } else {
                    int qrow = q0 + r + 8 * hi;
                    int kcol = j0 + nt * 16 + lm;
                    ok = mask[((size_t)b * Sq + qrow) * Skv + kcol] != 0;
                }
                if (!ok) sc[nt][r] = -1e9f;
            }
        }

        // online softmax (row stats replicated across each 16-lane half)
        float fs[8];
#pragma unroll
        for (int r = 0; r < 8; ++r) {
            float bm = fmaxf(sc[0][r], sc[1][r]);
            bm = fmaxf(bm, __shfl_xor(bm, 1, 32));
            bm = fmaxf(bm, __shfl_xor(bm, 2, 32));
            bm = fmaxf(bm, __shfl_xor(bm, 4, 32));
            bm = fmaxf(bm, __shfl_xor(bm, 8, 32));
            float mnew = fmaxf(mrow[r], bm);
            fs[r] = expf(mrow[r] - mnew);
            float p0 = expf(sc[0][r] - mnew);
            float p1 = expf(sc[1][r] - mnew);
            float r2 = p0 + p1;
            r2 += __shfl_xor(r2, 1, 32);
            r2 += __shfl_xor(r2, 2, 32);
            r2 += __shfl_xor(r2, 4, 32);
            r2 += __shfl_xor(r2, 8, 32);
            mrow[r] = mnew;
            lrow[r] = lrow[r] * fs[r] + r2;
            // scatter P (C-layout coords) into LDS for A-fragment re-read
            Ps[r + 8 * hi][lm]      = (_Float16)p0;
            Ps[r + 8 * hi][16 + lm] = (_Float16)p1;
        }
#pragma unroll
        for (int t = 0; t < 8; ++t)
#pragma unroll
            for (int r = 0; r < 8; ++r) o[t][r] *= fs[r];
        __syncthreads();

        // O += P(16x32) @ V(32x128): 8 output subtiles
        v16h pf;
#pragma unroll
        for (int p = 0; p < 8; ++p) {
            int kk = frag_k(p, hi);
            pf[2 * p]     = Ps[lm][kk];
            pf[2 * p + 1] = Ps[lm][kk + 1];
        }
#pragma unroll
        for (int t = 0; t < 8; ++t) {
            v16h bf;
#pragma unroll
            for (int p = 0; p < 8; ++p) {
                int kk = frag_k(p, hi);
                bf[2 * p]     = Vs[kk][t * 16 + lm];
                bf[2 * p + 1] = Vs[kk + 1][t * 16 + lm];
            }
            o[t] = wmma16(pf, bf, o[t]);
        }
        __syncthreads();
    }

    _Float16* Ob = O + ((size_t)b * Sq + q0) * Dm + (size_t)h * DSd;
#pragma unroll
    for (int r = 0; r < 8; ++r) {
        float inv = 1.f / lrow[r];
#pragma unroll
        for (int t = 0; t < 8; ++t)
            Ob[(size_t)(r + 8 * hi) * Dm + t * 16 + lm] = (_Float16)(o[t][r] * inv);
    }
}

// ---------------------------------------------------------------- launch
extern "C" void kernel_launch(void* const* d_in, const int* in_sizes, int n_in,
                              void* d_out, int out_size, void* d_ws, size_t ws_size,
                              hipStream_t stream) {
    (void)in_sizes; (void)n_in; (void)out_size; (void)ws_size;
    const int B = 16, S = 1024, D = 512, H = 4, SB = 256;
    const int BS = B * S;       // 16384 tokens
    const int BSB = B * SB;     // 4096 bb tokens

    const float* edges = (const float*)d_in[0];
    const float* bb    = (const float*)d_in[1];
    const unsigned char* gmask = (const unsigned char*)d_in[2];  // (B,32,32) bool
    const unsigned char* cmask = (const unsigned char*)d_in[3];  // (B,S,SB) bool
    const float* gWq = (const float*)d_in[4];  const float* gbq = (const float*)d_in[5];
    const float* gWk = (const float*)d_in[6];  const float* gbk = (const float*)d_in[7];
    const float* gWv = (const float*)d_in[8];  const float* gbv = (const float*)d_in[9];
    const float* gWo = (const float*)d_in[10]; const float* gbo = (const float*)d_in[11];
    const float* cWq = (const float*)d_in[12]; const float* cbq = (const float*)d_in[13];
    const float* cWk = (const float*)d_in[14]; const float* cbk = (const float*)d_in[15];
    const float* cWv = (const float*)d_in[16]; const float* cbv = (const float*)d_in[17];
    const float* cWo = (const float*)d_in[18]; const float* cbo = (const float*)d_in[19];
    const float* fW1 = (const float*)d_in[20]; const float* fb1 = (const float*)d_in[21];
    const float* fW2 = (const float*)d_in[22]; const float* fb2 = (const float*)d_in[23];

    char* ws = (char*)d_ws;
    size_t off = 0;
    auto alloc = [&](size_t bytes) -> char* {
        char* p = ws + off;
        off += (bytes + 255) & ~(size_t)255;
        return p;
    };
    // q16/k16/v16/ao16 form one contiguous 64 MB region; h16 (FFN hidden,
    // BS x 2048 f16 = 64 MB) aliases it since q/k/v/ao are dead by phase 3.
    _Float16* xn16 = (_Float16*)alloc((size_t)BS * D * 2);   // 16 MB
    _Float16* q16  = (_Float16*)alloc((size_t)BS * D * 2);   // 16 MB
    _Float16* k16  = (_Float16*)alloc((size_t)BS * D * 2);   // 16 MB
    _Float16* v16  = (_Float16*)alloc((size_t)BS * D * 2);   // 16 MB
    _Float16* ao16 = (_Float16*)alloc((size_t)BS * D * 2);   // 16 MB
    _Float16* h16  = q16;
    float* cur  = (float*)alloc((size_t)BS * D * 4);         // 32 MB
    float* cur2 = (float*)alloc((size_t)BS * D * 4);         // 32 MB
    _Float16* gWq16 = (_Float16*)alloc((size_t)D * D * 2);
    _Float16* gWk16 = (_Float16*)alloc((size_t)D * D * 2);
    _Float16* gWv16 = (_Float16*)alloc((size_t)D * D * 2);
    _Float16* gWo16 = (_Float16*)alloc((size_t)D * D * 2);
    _Float16* cWq16 = (_Float16*)alloc((size_t)D * D * 2);
    _Float16* cWk16 = (_Float16*)alloc((size_t)D * D * 2);
    _Float16* cWv16 = (_Float16*)alloc((size_t)D * D * 2);
    _Float16* cWo16 = (_Float16*)alloc((size_t)D * D * 2);
    _Float16* fW116 = (_Float16*)alloc((size_t)4 * D * D * 2);
    _Float16* fW216 = (_Float16*)alloc((size_t)4 * D * D * 2);
    _Float16* bb16  = (_Float16*)alloc((size_t)BSB * D * 2);

    auto conv = [&](const float* s, _Float16* dcv, int n) {
        f32_to_f16_kernel<<<(n + 255) / 256, 256, 0, stream>>>(s, dcv, n);
    };
    conv(gWq, gWq16, D * D); conv(gWk, gWk16, D * D);
    conv(gWv, gWv16, D * D); conv(gWo, gWo16, D * D);
    conv(cWq, cWq16, D * D); conv(cWk, cWk16, D * D);
    conv(cWv, cWv16, D * D); conv(cWo, cWo16, D * D);
    conv(fW1, fW116, 4 * D * D); conv(fW2, fW216, 4 * D * D);
    conv(bb, bb16, BSB * D);

    const float qscale = 0.08838834764831845f;  // 1/sqrt(128)
    const dim3 blk(128);

    // ---- Phase 1: masked global self-attention
    inorm512_kernel<<<BS, 256, 0, stream>>>(edges, xn16);
    gemm_xwt_kernel<0, 0><<<dim3(D / 16, BS / 64), blk, 0, stream>>>(
        xn16, gWq16, gbq, nullptr, nullptr, q16, BS, D, D, qscale);
    gemm_xwt_kernel<0, 0><<<dim3(D / 16, BS / 64), blk, 0, stream>>>(
        xn16, gWk16, gbk, nullptr, nullptr, k16, BS, D, D, 1.f);
    gemm_xwt_kernel<0, 0><<<dim3(D / 16, BS / 64), blk, 0, stream>>>(
        xn16, gWv16, gbv, nullptr, nullptr, v16, BS, D, D, 1.f);
    flash_attn_kernel<1><<<dim3(S / 16, H, B), 32, 0, stream>>>(
        q16, k16, v16, gmask, ao16, S, S);
    gemm_xwt_kernel<0, 1><<<dim3(D / 16, BS / 64), blk, 0, stream>>>(
        ao16, gWo16, gbo, edges, cur, nullptr, BS, D, D, 1.f);

    // ---- Phase 2: cross-attention from bb semantics
    inorm512_kernel<<<BS, 256, 0, stream>>>(cur, xn16);
    gemm_xwt_kernel<0, 0><<<dim3(D / 16, BS / 64), blk, 0, stream>>>(
        xn16, cWq16, cbq, nullptr, nullptr, q16, BS, D, D, qscale);
    gemm_xwt_kernel<0, 0><<<dim3(D / 16, BSB / 64), blk, 0, stream>>>(
        bb16, cWk16, cbk, nullptr, nullptr, k16, BSB, D, D, 1.f);
    gemm_xwt_kernel<0, 0><<<dim3(D / 16, BSB / 64), blk, 0, stream>>>(
        bb16, cWv16, cbv, nullptr, nullptr, v16, BSB, D, D, 1.f);
    flash_attn_kernel<0><<<dim3(S / 16, H, B), 32, 0, stream>>>(
        q16, k16, v16, cmask, ao16, S, SB);
    gemm_xwt_kernel<0, 1><<<dim3(D / 16, BS / 64), blk, 0, stream>>>(
        ao16, cWo16, cbo, cur, cur2, nullptr, BS, D, D, 1.f);

    // ---- Phase 3: FFN
    inorm512_kernel<<<BS, 256, 0, stream>>>(cur2, xn16);
    gemm_xwt_kernel<1, 0><<<dim3(4 * D / 16, BS / 64), blk, 0, stream>>>(
        xn16, fW116, fb1, nullptr, nullptr, h16, BS, 4 * D, D, 1.f);
    gemm_xwt_kernel<0, 1><<<dim3(D / 16, BS / 64), blk, 0, stream>>>(
        h16, fW216, fb2, cur2, (float*)d_out, nullptr, BS, D, 4 * D, 1.f);
}
